// EBmRNN_61950608278070
// MI455X (gfx1250) — matrix operand order repaired and checked by reading, not compiled, not run on hardware
//
#include <hip/hip_runtime.h>
#include <stdint.h>

// ---------------------------------------------------------------------------
// EBmRNN for MI455X (gfx1250): persistent batch-tiled scan.
//   grid = 4 workgroups (16 batch rows each), 256 threads = 8 wave32.
//   All GEMMs via v_wmma_f32_16x16x32_bf16 (bf16 in, f32 accum).
//   A fragments hoisted to registers per phase; gi/gh GEMMs interleaved as two
//   accumulator chains to hide L2 weight-load latency behind WMMA issue.
//   Recurrent state in LDS; em/bm memories + bf16 weights in d_ws (L2-hot).
// ---------------------------------------------------------------------------

typedef __attribute__((ext_vector_type(16))) __bf16 v16bf;
typedef __attribute__((ext_vector_type(8)))  __bf16 v8bf;
typedef __attribute__((ext_vector_type(8)))  float  v8f;

union B16x8 { uint4 u; v8bf v; };

__device__ __forceinline__ uint16_t f2bf(float f) {
  uint32_t u = __float_as_uint(f);
  uint32_t r = (u + 0x7FFFu + ((u >> 16) & 1u)) >> 16;   // RNE
  return (uint16_t)r;
}
__device__ __forceinline__ float sigm(float x) { return 1.f / (1.f + __expf(-x)); }

// --- hoist the 16x256 bf16 A operand (LDS) into 8 k-fragments ---------------
__device__ __forceinline__ void load_afrag(const uint16_t* sAa, int lane, v16bf A[8]) {
  const int m  = lane & 15;
  const int kb = (lane >> 4) << 4;                        // 0 / 16
  const uint16_t* ap = sAa + m * 256 + kb;
#pragma unroll
  for (int kt = 0; kt < 8; ++kt) {
    B16x8 a0, a1;
    a0.u = *(const uint4*)(ap + kt * 32);
    a1.u = *(const uint4*)(ap + kt * 32 + 8);
    A[kt] = __builtin_shufflevector(a0.v, a1.v, 0,1,2,3,4,5,6,7,8,9,10,11,12,13,14,15);
  }
}

// --- two independent 16x16 tiles, interleaved WMMA chains --------------------
__device__ __forceinline__ void gemm_tile_dual(
    const uint16_t* __restrict__ W0, const v16bf* A0, float* sO0, int jt0,
    const uint16_t* __restrict__ W1, const v16bf* A1, float* sO1, int jt1,
    int outStride, int lane) {
  const int n   = lane & 15;
  const int klo = (lane >> 4) << 4;                       // 0 / 16
  const uint16_t* bp0 = W0 + (jt0 * 16 + n) * 256 + klo;
  const uint16_t* bp1 = W1 + (jt1 * 16 + n) * 256 + klo;
  v8f acc0 = {0.f,0.f,0.f,0.f,0.f,0.f,0.f,0.f};
  v8f acc1 = {0.f,0.f,0.f,0.f,0.f,0.f,0.f,0.f};
#pragma unroll
  for (int kt = 0; kt < 8; ++kt) {                        // K = 256 = 8*32
    B16x8 b0, b1, c0, c1;
    b0.u = *(const uint4*)(bp0 + kt * 32);
    b1.u = *(const uint4*)(bp0 + kt * 32 + 8);
    c0.u = *(const uint4*)(bp1 + kt * 32);
    c1.u = *(const uint4*)(bp1 + kt * 32 + 8);
    v16bf B0 = __builtin_shufflevector(b0.v, b1.v, 0,1,2,3,4,5,6,7,8,9,10,11,12,13,14,15);
    v16bf B1 = __builtin_shufflevector(c0.v, c1.v, 0,1,2,3,4,5,6,7,8,9,10,11,12,13,14,15);
    acc0 = __builtin_amdgcn_wmma_f32_16x16x32_bf16(false, A0[kt], false, B0,
                                                   (short)0, acc0, false, false);
    acc1 = __builtin_amdgcn_wmma_f32_16x16x32_bf16(false, A1[kt], false, B1,
                                                   (short)0, acc1, false, false);
  }
  const int mb = (lane >> 4) << 3;
  float* o0 = sO0 + jt0 * 16 + n;
  float* o1 = sO1 + jt1 * 16 + n;
#pragma unroll
  for (int r = 0; r < 8; ++r) {
    o0[(mb + r) * outStride] = acc0[r];
    o1[(mb + r) * outStride] = acc1[r];
  }
}

// --- single 16x16 tile (preloaded A) -----------------------------------------
__device__ __forceinline__ void gemm_tile(const uint16_t* __restrict__ Wb,
                                          const v16bf* A, float* sOut,
                                          int outStride, int jtile, int lane) {
  const int n   = lane & 15;
  const int klo = (lane >> 4) << 4;
  const uint16_t* bp = Wb + (jtile * 16 + n) * 256 + klo;
  v8f acc = {0.f,0.f,0.f,0.f,0.f,0.f,0.f,0.f};
#pragma unroll
  for (int kt = 0; kt < 8; ++kt) {
    B16x8 b0, b1;
    b0.u = *(const uint4*)(bp + kt * 32);
    b1.u = *(const uint4*)(bp + kt * 32 + 8);
    v16bf Bv = __builtin_shufflevector(b0.v, b1.v, 0,1,2,3,4,5,6,7,8,9,10,11,12,13,14,15);
    acc = __builtin_amdgcn_wmma_f32_16x16x32_bf16(false, A[kt], false, Bv,
                                                  (short)0, acc, false, false);
  }
  const int mb = (lane >> 4) << 3;
  float* op = sOut + jtile * 16 + n;
#pragma unroll
  for (int r = 0; r < 8; ++r) op[(mb + r) * outStride] = acc[r];
}

// --- fp32 -> bf16 weight repack (runs once per launch, on stream) ------------
__global__ void convert_w(const float* __restrict__ a0, const float* __restrict__ a1,
                          const float* __restrict__ a2, const float* __restrict__ a3,
                          const float* __restrict__ a4, uint16_t* __restrict__ o) {
  int i = blockIdx.x * 256 + threadIdx.x;
  if (i >= 868352) return;
  float v;
  if      (i < 196608) v = a0[i];
  else if (i < 393216) v = a1[i - 196608];
  else if (i < 589824) v = a2[i - 393216];
  else if (i < 786432) v = a3[i - 589824];
  else                 v = a4[i - 786432];
  o[i] = f2bf(v);
}

#define TB 256

__global__ __launch_bounds__(256, 1)
void ebm_rnn_kernel(const float* __restrict__ x,      // [64][512][128]
                    const float* __restrict__ h0in,   // [2][64][256]
                    const float* __restrict__ bih0, const float* __restrict__ bhh0,
                    const float* __restrict__ bih1, const float* __restrict__ bhh1,
                    const float* __restrict__ bint,
                    const float* __restrict__ wmix,   // [32][64]
                    const float* __restrict__ bmix,   // [32]
                    uint8_t* __restrict__ ws,
                    float* __restrict__ out) {        // [64][512][640]
  // ---- LDS carve (155,648 B total; phase-aliased) ----
  __shared__ __align__(16) uint8_t smem[155648];
  uint16_t* sA  = (uint16_t*)(smem);            // [16][256] bf16  A-operand
  uint16_t* sHb = (uint16_t*)(smem + 8192);     // [16][256] bf16  h_prev operand
  float* sH0 = (float*)(smem + 16384);          // [16][256] f32 persistent
  float* sH1 = (float*)(smem + 32768);          // [16][256]
  float* sLR = (float*)(smem + 49152);          // [16][128] l_read persistent
  float* sGi = (float*)(smem + 57344);          // [16][768] gate buf i
  float* sItf = sGi;                            // alias: [16][320] interface
  float* sER = (float*)(smem + 77824);          // [16][4][32] em_read
  float* sBR = (float*)(smem + 86016);          // [16][4][32] bm_read
  float* sEr = (float*)(smem + 94208);          // [16][32] em_erased
  float* sEs = (float*)(smem + 96256);          // [16][32] sigmoid(e_t)
  float* sGh = (float*)(smem + 106496);         // [16][768] gate buf h
  float* sNE = sGh;                             // alias: [16][64] em row norms
  float* sNB  = (float*)(smem + 110592);        // [16][64] bm row norms
  float* sS1  = (float*)(smem + 114688);        // [64][64] em scores/probs
  float* sS2  = (float*)(smem + 131072);        // [64][64] bm scores/probs
  float* sNK1 = (float*)(smem + 147456);        // [64] |k_em|+eps
  float* sNK2 = (float*)(smem + 147712);        // [64] |k_bm|+eps

  const int tid  = threadIdx.x;
  const int lane = tid & 31;
  const int wave = tid >> 5;
  const int bt   = blockIdx.x;                  // batch tile: rows bt*16..bt*16+15

  const uint16_t* Wih0 = (const uint16_t*)ws;
  const uint16_t* Whh0 = Wih0 + 196608;
  const uint16_t* Wih1 = Wih0 + 393216;
  const uint16_t* Whh1 = Wih0 + 589824;
  const uint16_t* Wint = Wih0 + 786432;
  float* em = (float*)(ws + 1736704)          + (size_t)bt * 16 * 64 * 32;
  float* bm = (float*)(ws + 1736704 + 524288) + (size_t)bt * 16 * 64 * 32;

  // ---- init state (fresh every launch: ws is not trusted across calls) ----
  for (int i = tid; i < 16 * 64 * 32; i += TB) { em[i] = 0.f; bm[i] = 0.f; }
  for (int i = tid; i < 16 * 256; i += TB) {
    int m = i >> 8, h = i & 255, b = bt * 16 + m;
    sH0[i] = h0in[b * 256 + h];
    sH1[i] = h0in[64 * 256 + b * 256 + h];
  }
  for (int i = tid; i < 16 * 128; i += TB) sLR[i] = 0.f;
  __syncthreads();

  v16bf Ai[8], Ah[8];

  for (int t = 0; t < 512; ++t) {
    // ---- P1: ctl = [x_t, l_read] -> bf16; stage h0_prev bf16 ----
    for (int i = tid; i < 16 * 256; i += TB) {
      int m = i >> 8, k = i & 255, b = bt * 16 + m;
      float v;
      if (k < 128) {
        const float* xp = &x[((size_t)b * 512 + t) * 128 + k];
        v = *xp;
        if (t < 511) __builtin_prefetch(xp + 128, 0, 3);  // next-step x_t
      } else {
        v = sLR[m * 128 + (k - 128)];
      }
      sA[i]  = f2bf(v);
      sHb[i] = f2bf(sH0[i]);
    }
    __syncthreads();

    // ---- GRU0 GEMMs: gi = ctl@Wih0^T || gh = h0@Whh0^T (dual chains) ----
    load_afrag(sA,  lane, Ai);
    load_afrag(sHb, lane, Ah);
#pragma unroll 1
    for (int j = 0; j < 6; ++j) {
      int jt = wave * 6 + j;
      gemm_tile_dual(Wih0, Ai, sGi, jt, Whh0, Ah, sGh, jt, 768, lane);
    }
    __syncthreads();

    // ---- GRU0 gates ----
    for (int i = tid; i < 16 * 256; i += TB) {
      int m = i >> 8, h = i & 255, ba = m * 768;
      float r  = sigm(sGi[ba + h]       + bih0[h]       + sGh[ba + h]       + bhh0[h]);
      float z  = sigm(sGi[ba + 256 + h] + bih0[256 + h] + sGh[ba + 256 + h] + bhh0[256 + h]);
      float nn = tanhf(sGi[ba + 512 + h] + bih0[512 + h] +
                       r * (sGh[ba + 512 + h] + bhh0[512 + h]));
      float hv = (1.f - z) * nn + z * sH0[i];
      sH0[i] = hv;
      sA[i]  = f2bf(hv);        // A-operand for GRU1
      sHb[i] = f2bf(sH1[i]);    // h1_prev operand
    }
    __syncthreads();

    // ---- GRU1 GEMMs ----
    load_afrag(sA,  lane, Ai);
    load_afrag(sHb, lane, Ah);
#pragma unroll 1
    for (int j = 0; j < 6; ++j) {
      int jt = wave * 6 + j;
      gemm_tile_dual(Wih1, Ai, sGi, jt, Whh1, Ah, sGh, jt, 768, lane);
    }
    __syncthreads();

    // ---- GRU1 gates ----
    for (int i = tid; i < 16 * 256; i += TB) {
      int m = i >> 8, h = i & 255, ba = m * 768;
      float r  = sigm(sGi[ba + h]       + bih1[h]       + sGh[ba + h]       + bhh1[h]);
      float z  = sigm(sGi[ba + 256 + h] + bih1[256 + h] + sGh[ba + 256 + h] + bhh1[256 + h]);
      float nn = tanhf(sGi[ba + 512 + h] + bih1[512 + h] +
                       r * (sGh[ba + 512 + h] + bhh1[512 + h]));
      float hv = (1.f - z) * nn + z * sH1[i];
      sH1[i] = hv;
      sA[i]  = f2bf(hv);        // A-operand for interface GEMM
    }
    __syncthreads();

    // ---- interface GEMM: itf = h1n @ Wint^T  (20 N-tiles: 8 dual + 4 single)
    load_afrag(sA, lane, Ai);
    gemm_tile_dual(Wint, Ai, sItf, wave, Wint, Ai, sItf, wave + 8, 320, lane);
    if (wave < 4) gemm_tile(Wint, Ai, sItf, 320, 16 + wave, lane);
    __syncthreads();
    for (int i = tid; i < 16 * 320; i += TB) {
      int j = i % 320;
      float v = sItf[i] + bint[j];
      sItf[i] = fminf(fmaxf(v, 0.f), 20.f);   // clip(relu(.), -20, 20)
    }
    __syncthreads();

    // ---- M1: snapshot old em[slot], write m_t, e = sigmoid(e_t) ----
    const int slot = t & 63;
    for (int i = tid; i < 512; i += TB) {
      int m = i >> 5, w = i & 31;
      float* ep = &em[(m * 64 + slot) * 32 + w];
      sEr[i] = *ep;
      *ep    = sItf[m * 320 + 256 + w];
      sEs[i] = sigm(sItf[m * 320 + 288 + w]);
    }
    __syncthreads();

    // ---- M2: row norms of em_new / bm, key norms ----
    for (int i = tid; i < 1024; i += TB) {
      int m = i >> 6, n = i & 63;
      const float* er_ = &em[(m * 64 + n) * 32];
      const float* br_ = &bm[(m * 64 + n) * 32];
      float se = 0.f, sb = 0.f;
#pragma unroll
      for (int w = 0; w < 32; ++w) { se += er_[w] * er_[w]; sb += br_[w] * br_[w]; }
      sNE[i] = sqrtf(se); sNB[i] = sqrtf(sb);
    }
    if (tid < 64) {
      int m = tid >> 2, r = tid & 3;
      const float* k1 = &sItf[m * 320 + r * 32];
      const float* k2 = &sItf[m * 320 + 128 + r * 32];
      float n1 = 0.f, n2 = 0.f;
#pragma unroll
      for (int w = 0; w < 32; ++w) { n1 += k1[w] * k1[w]; n2 += k2[w] * k2[w]; }
      sNK1[tid] = sqrtf(n1) + 1e-8f;
      sNK2[tid] = sqrtf(n2) + 1e-8f;
    }
    __syncthreads();

    // ---- M3a: cosine scores (m,r,n) ----
    for (int i = tid; i < 4096; i += TB) {
      int mr = i >> 6, n = i & 63, m = mr >> 2, r = mr & 3;
      const float* k1 = &sItf[m * 320 + r * 32];
      const float* k2 = &sItf[m * 320 + 128 + r * 32];
      const float* Mr = &em[(m * 64 + n) * 32];
      const float* Br = &bm[(m * 64 + n) * 32];
      float d1 = 0.f, d2 = 0.f;
#pragma unroll
      for (int w = 0; w < 32; ++w) { d1 += k1[w] * Mr[w]; d2 += k2[w] * Br[w]; }
      sS1[i] = d1 / (sNK1[mr] * (sNE[m * 64 + n] + 1e-8f)) * (1.f / 0.3f);
      sS2[i] = d2 / (sNK2[mr] * (sNB[m * 64 + n] + 1e-8f));
    }
    __syncthreads();

    // ---- M3b: softmax over n (per head), in place ----
    if (tid < 64) {
      float* p1 = &sS1[tid * 64];
      float* p2 = &sS2[tid * 64];
      float mx1 = -3.4e38f, mx2 = -3.4e38f;
      for (int n = 0; n < 64; ++n) { mx1 = fmaxf(mx1, p1[n]); mx2 = fmaxf(mx2, p2[n]); }
      float Z1 = 0.f, Z2 = 0.f;
      for (int n = 0; n < 64; ++n) {
        float e1 = __expf(p1[n] - mx1); p1[n] = e1; Z1 += e1;
        float e2 = __expf(p2[n] - mx2); p2[n] = e2; Z2 += e2;
      }
      float i1 = 1.f / Z1, i2 = 1.f / Z2;
      for (int n = 0; n < 64; ++n) { p1[n] *= i1; p2[n] *= i2; }
    }
    __syncthreads();

    // ---- M3c: weighted reads (m,r,w) ----
    for (int i = tid; i < 2048; i += TB) {
      int mr = i >> 5, w = i & 31, m = mr >> 2;
      const float* p1 = &sS1[mr * 64];
      const float* p2 = &sS2[mr * 64];
      float er = 0.f, br = 0.f;
      for (int n = 0; n < 64; ++n) {
        er += p1[n] * em[(m * 64 + n) * 32 + w];
        br += p2[n] * bm[(m * 64 + n) * 32 + w];
      }
      sER[mr * 32 + w] = er; sBR[mr * 32 + w] = br;
    }
    __syncthreads();

    // ---- M4: bm update (deterministic 4-term ww mean) + mix gate ----
    for (int i = tid; i < 1024; i += TB) {
      int m = i >> 6, n = i & 63;
      float wv = 0.25f * (sS2[(m * 4 + 0) * 64 + n] + sS2[(m * 4 + 1) * 64 + n] +
                          sS2[(m * 4 + 2) * 64 + n] + sS2[(m * 4 + 3) * 64 + n]);
      float* br_ = &bm[(m * 64 + n) * 32];
#pragma unroll
      for (int w = 0; w < 32; ++w)
        br_[w] = br_[w] * (1.f - wv * sEs[m * 32 + w]) + wv * sEr[m * 32 + w];
    }
    for (int i = tid; i < 2048; i += TB) {
      int mr = i >> 5, w = i & 31, m = mr >> 2, r = mr & 3;
      const float* cE = &sER[mr * 32];
      const float* cB = &sBR[mr * 32];
      float acc = bmix[w];
#pragma unroll
      for (int k = 0; k < 32; ++k)
        acc += wmix[w * 64 + k] * cE[k] + wmix[w * 64 + 32 + k] * cB[k];
      float g = sigm(acc);
      sLR[m * 128 + r * 32 + w] = g * cE[w] + (1.f - g) * cB[w];
    }
    __syncthreads();

    // ---- output: [h0n | h1n | l_new] ----
    for (int i = tid; i < 16 * 640; i += TB) {
      int m = i / 640, c = i - m * 640;
      float v = (c < 256) ? sH0[m * 256 + c]
              : (c < 512) ? sH1[m * 256 + (c - 256)]
                          : sLR[m * 128 + (c - 512)];
      out[(((size_t)(bt * 16 + m)) * 512 + t) * 640 + c] = v;
    }
    __syncthreads();
  }
}

extern "C" void kernel_launch(void* const* d_in, const int* in_sizes, int n_in,
                              void* d_out, int out_size, void* d_ws, size_t ws_size,
                              hipStream_t stream) {
  (void)in_sizes; (void)n_in; (void)out_size; (void)ws_size;
  const float* x    = (const float*)d_in[0];
  const float* h0   = (const float*)d_in[1];
  const float* Wih0 = (const float*)d_in[2];
  const float* Whh0 = (const float*)d_in[3];
  const float* bih0 = (const float*)d_in[4];
  const float* bhh0 = (const float*)d_in[5];
  const float* Wih1 = (const float*)d_in[6];
  const float* Whh1 = (const float*)d_in[7];
  const float* bih1 = (const float*)d_in[8];
  const float* bhh1 = (const float*)d_in[9];
  const float* Wint = (const float*)d_in[10];
  const float* bint = (const float*)d_in[11];
  const float* Wmix = (const float*)d_in[12];
  const float* bmix = (const float*)d_in[13];

  convert_w<<<(868352 + 255) / 256, 256, 0, stream>>>(Wih0, Whh0, Wih1, Whh1, Wint,
                                                      (uint16_t*)d_ws);
  ebm_rnn_kernel<<<4, 256, 0, stream>>>(x, h0, bih0, bhh0, bih1, bhh1, bint,
                                        Wmix, bmix, (uint8_t*)d_ws, (float*)d_out);
}
